// CRFLayer_51427938402390
// MI455X (gfx1250) — compile-verified, matrix operand order checked by reading, not run on hardware
//
#include <hip/hip_runtime.h>
#include <hip/hip_bf16.h>

typedef __attribute__((ext_vector_type(16))) _Float16 v16h;
typedef __attribute__((ext_vector_type(8)))  float    v8f;

#define B_ 256
#define S_ 512
#define T_ 128

// ---------------------------------------------------------------------------
// Numerator: path score per batch. One wave32 per batch; lanes stride over S.
// ---------------------------------------------------------------------------
__global__ __launch_bounds__(32) void crf_numerator(
    const float* __restrict__ em, const int* __restrict__ tags,
    const unsigned char* __restrict__ mask, const float* __restrict__ startT,
    const float* __restrict__ endT, const float* __restrict__ trans,
    float* __restrict__ ws_num) {
  const int b = blockIdx.x;
  const int lane = threadIdx.x;
  const int* tb = tags + b * S_;
  const unsigned char* mb = mask + b * S_;
  const float* eb = em + (long)b * (S_ * T_);
  float acc = 0.f;
  int cnt = 0;
  for (int s = lane; s < S_; s += 32) {
    const int m = mb[s] ? 1 : 0;
    cnt += m;
    if (s > 0 && m) {
      const int tp = tb[s - 1], tc = tb[s];
      acc += trans[tp * T_ + tc] + eb[s * T_ + tc];
    }
  }
  for (int off = 16; off > 0; off >>= 1) {
    acc += __shfl_xor(acc, off, 32);
    cnt += __shfl_xor(cnt, off, 32);
  }
  if (lane == 0) {
    const int t0 = tb[0];
    float sc = startT[t0] + eb[t0] + acc;
    const int send = cnt - 1;                 // seq_ends = sum(mask) - 1
    sc += endT[tb[send]];
    ws_num[b] = sc;
  }
}

// ---------------------------------------------------------------------------
// Forward algorithm in exp-domain with per-step renormalization.
// 16 blocks x 256 threads (8 waves). Block handles batches [16*blk, 16*blk+16).
// Wave w owns output columns j in [16w, 16w+16); keeps B = exp(trans) tile
// resident in registers; per step does 4x v_wmma_f32_16x16x32_f16 (K=128).
// Row max via LDS segment pass (no ds_bpermute chains); one v_rcp_f32 per row
// per step, applied scale folded exactly into carried log-correction c.
// ---------------------------------------------------------------------------
__global__ __launch_bounds__(256) void crf_forward(
    const float* __restrict__ em, const unsigned char* __restrict__ mask,
    const float* __restrict__ startT, const float* __restrict__ endT,
    const float* __restrict__ trans, float* __restrict__ ws_den) {
  __shared__ __align__(16) float    u_lds[16][132];   // unnormalized alpha (f32)
  __shared__ __align__(16) _Float16 p_lds[16][136];   // normalized probs (f16)
  __shared__ __align__(16) float    seg_lds[16][16];  // per-(row,segment) maxima
  __shared__ float part[8][16];                       // final-reduction partials
  __shared__ float inv_lds[16];                       // per-row scale (rcp of max)
  __shared__ float c_lds[16];                         // carried log-scale
  __shared__ float mk_lds[16];                        // mask for current step

  const int tid   = threadIdx.x;
  const int lane  = tid & 31;
  const int w     = tid >> 5;        // wave 0..7
  const int j0    = w * 16;          // output column tile base
  const int N     = lane & 15;       // column within tile (B/C/D layouts)
  const int hi    = lane >> 4;       // half-wave id
  const int khalf = hi * 8;          // K-half offset for 16-bit A/B layouts
  const int b0    = blockIdx.x * 16; // batch tile base

  // --- Load B-matrix fragments: E = exp(trans), resident for whole scan. ---
  // 16-bit 32x16 layout (mirrors ISA 16x32 A layout with N in place of M):
  // lane = N (mod 16); elems 0..7 -> K = 32k + khalf + e ; 8..15 -> +16.
  v16h bfrag[4];
  #pragma unroll
  for (int k = 0; k < 4; ++k) {
    #pragma unroll
    for (int e = 0; e < 16; ++e) {
      const int K = 32 * k + khalf + ((e < 8) ? e : (16 + (e - 8)));
      bfrag[k][e] = (_Float16)__expf(trans[K * T_ + (j0 + N)]);
    }
  }

  if (tid < 16) c_lds[tid] = 0.f;

  // --- Init: u = exp(start_j + em[b,0,j]); C-layout frags (VGPR v -> row v+8*hi). ---
  float ufrag[8];
  #pragma unroll
  for (int v = 0; v < 8; ++v) {
    const int bm = v + 8 * hi;
    const int j = j0 + N;
    ufrag[v] = __expf(startT[j] + em[(long)(b0 + bm) * (S_ * T_) + j]);
  }

  for (int s = 1; s < S_; ++s) {
    // (A) stage u tiles to LDS
    #pragma unroll
    for (int v = 0; v < 8; ++v) u_lds[v + 8 * hi][j0 + N] = ufrag[v];
    __syncthreads();

    // (C) segment max: thread t covers row t>>4, columns [8*(t&15), +8)
    {
      const int row = tid >> 4, seg = tid & 15;
      const float4 a4 = *reinterpret_cast<const float4*>(&u_lds[row][seg * 8]);
      const float4 b4 = *reinterpret_cast<const float4*>(&u_lds[row][seg * 8 + 4]);
      const float m8 = fmaxf(fmaxf(fmaxf(a4.x, a4.y), fmaxf(a4.z, a4.w)),
                             fmaxf(fmaxf(b4.x, b4.y), fmaxf(b4.z, b4.w)));
      seg_lds[row][seg] = m8;
    }
    __syncthreads();

    // (E) combine 16 segment maxima -> one v_rcp_f32 per row; fold scale into c
    if (tid < 16) {
      const float4* sp = reinterpret_cast<const float4*>(&seg_lds[tid][0]);
      const float4 q0 = sp[0], q1 = sp[1], q2 = sp[2], q3 = sp[3];
      const float mx =
          fmaxf(fmaxf(fmaxf(fmaxf(q0.x, q0.y), fmaxf(q0.z, q0.w)),
                      fmaxf(fmaxf(q1.x, q1.y), fmaxf(q1.z, q1.w))),
                fmaxf(fmaxf(fmaxf(q2.x, q2.y), fmaxf(q2.z, q2.w)),
                      fmaxf(fmaxf(q3.x, q3.y), fmaxf(q3.z, q3.w))));
      const float inv = __builtin_amdgcn_rcpf(mx);   // v_rcp_f32 (~1 ulp)
      inv_lds[tid] = inv;
      c_lds[tid] -= __logf(inv);   // exact compensation of the applied scale
      mk_lds[tid] = mask[(b0 + tid) * S_ + s] ? 1.f : 0.f;
    }
    __syncthreads();

    // (G) normalize: p = (f16)(u * inv); paired -> b64 load, pk cvt, b32 store
    #pragma unroll
    for (int r = 0; r < 4; ++r) {
      const int q   = r * 256 + tid;   // 0..1023 column pairs
      const int row = q >> 6;          // /64 pairs per row
      const int c2  = (q & 63) * 2;
      const float2 uv = *reinterpret_cast<const float2*>(&u_lds[row][c2]);
      const float inv = inv_lds[row];
      union { _Float16 h[2]; unsigned int u32; } pk;
      pk.h[0] = (_Float16)(uv.x * inv);
      pk.h[1] = (_Float16)(uv.y * inv);
      *reinterpret_cast<unsigned int*>(&p_lds[row][c2]) = pk.u32;
    }
    __syncthreads();

    // (I) GEMM: dot_j = sum_i p_i * E_ij  (4 chained WMMAs over K=128)
    v8f acc = {};
    #pragma unroll
    for (int k = 0; k < 4; ++k) {
      // 16-bit 16x32 A layout: lane -> M = lane%16; dword pairs of K values.
      union { v16h h; unsigned int u32[8]; } a;
      const int kb = 32 * k + khalf;
      #pragma unroll
      for (int p = 0; p < 4; ++p) {
        a.u32[p]     = *reinterpret_cast<const unsigned int*>(&p_lds[N][kb + 2 * p]);
        a.u32[p + 4] = *reinterpret_cast<const unsigned int*>(&p_lds[N][kb + 16 + 2 * p]);
      }
      acc = __builtin_amdgcn_wmma_f32_16x16x32_f16(
          /*neg_a=*/false, a.h, /*neg_b=*/false, bfrag[k],
          /*c_mod=*/(short)0, acc, /*reuse_a=*/false, /*reuse_b=*/false);
    }

    // u' = mask ? dot * exp(em) : u * inv   (scale already folded into c)
    #pragma unroll
    for (int v = 0; v < 8; ++v) {
      const int bm = v + 8 * hi;
      const long eidx = (long)(b0 + bm) * (S_ * T_) + (long)s * T_ + (j0 + N);
      const float ee  = __expf(em[eidx]);
      const float mkv = mk_lds[bm];
      const float un  = acc[v] * ee;
      const float uo  = ufrag[v] * inv_lds[bm];
      ufrag[v] = mkv * un + (1.f - mkv) * uo;
      if (s + 1 < S_) __builtin_prefetch(&em[eidx + T_], 0, 1);  // next step
    }
    // no trailing barrier: next iteration's barriers order all LDS hazards
  }

  // --- denominator_b = c_b + log( sum_j u_j * exp(end_j) ) ---
  const float ee = __expf(endT[j0 + N]);
  #pragma unroll
  for (int v = 0; v < 8; ++v) {
    const int bm = v + 8 * hi;
    float sv = ufrag[v] * ee;
    sv += __shfl_xor(sv, 1, 16);
    sv += __shfl_xor(sv, 2, 16);
    sv += __shfl_xor(sv, 4, 16);
    sv += __shfl_xor(sv, 8, 16);
    if (N == 0) part[w][bm] = sv;
  }
  __syncthreads();
  if (tid < 16) {
    float tot = 0.f;
    #pragma unroll
    for (int q = 0; q < 8; ++q) tot += part[q][tid];
    ws_den[b0 + tid] = c_lds[tid] + __logf(tot);
  }
}

// ---------------------------------------------------------------------------
// Deterministic final reduction: out = sum_b (den_b - num_b)
// ---------------------------------------------------------------------------
__global__ __launch_bounds__(256) void crf_reduce(
    const float* __restrict__ ws_num, const float* __restrict__ ws_den,
    float* __restrict__ out) {
  __shared__ float sm[256];
  const int t = threadIdx.x;
  sm[t] = ws_den[t] - ws_num[t];
  __syncthreads();
  for (int off = 128; off > 0; off >>= 1) {
    if (t < off) sm[t] += sm[t + off];
    __syncthreads();
  }
  if (t == 0) out[0] = sm[0];
}

extern "C" void kernel_launch(void* const* d_in, const int* in_sizes, int n_in,
                              void* d_out, int out_size, void* d_ws, size_t ws_size,
                              hipStream_t stream) {
  const float*         em     = (const float*)d_in[0];
  const int*           tags   = (const int*)d_in[1];
  const unsigned char* mask   = (const unsigned char*)d_in[2];
  const float*         startT = (const float*)d_in[3];
  const float*         endT   = (const float*)d_in[4];
  const float*         trans  = (const float*)d_in[5];
  float* ws_num = (float*)d_ws;        // 256 floats
  float* ws_den = ws_num + B_;         // 256 floats

  crf_numerator<<<B_, 32, 0, stream>>>(em, tags, mask, startT, endT, trans, ws_num);
  crf_forward<<<B_ / 16, 256, 0, stream>>>(em, mask, startT, endT, trans, ws_den);
  crf_reduce<<<1, 256, 0, stream>>>(ws_num, ws_den, (float*)d_out);
}
